// SRUModel_58918361366550
// MI455X (gfx1250) — matrix operand order repaired and checked by reading, not compiled, not run on hardware
//
#include <hip/hip_runtime.h>
#include <hip/hip_bf16.h>
#include <stdint.h>

typedef __bf16 bf16;
typedef bf16  v8bf  __attribute__((ext_vector_type(8)));
typedef bf16  v16bf __attribute__((ext_vector_type(16)));
typedef float v8f   __attribute__((ext_vector_type(8)));
typedef int   v4i   __attribute__((ext_vector_type(4)));

#define VOCAB 32000
#define N_E   1024
#define N_D   1024
#define DEPTH 4
#define SEQ_L 512
#define BATCH 8
#define M_TOT (SEQ_L * BATCH)   /* 4096 rows of the activation matrix */

/* GEMM tiling: block = 128x128, K-chunk 64, 8 wave32 waves (4x2),
 * wave tile = 32x64 = 2x4 accumulators of v_wmma_f32_16x16x32_bf16.
 * Double-buffered LDS (2 x 32 KB of the 320 KB WGP pool). */
#define BM 128
#define BN 128
#define BK 64

// CDNA5 async global->LDS path (ASYNCcnt) when the toolchain exposes it;
// otherwise fall back to register-staged double buffering.
#if defined(__HIP_DEVICE_COMPILE__) && \
    __has_builtin(__builtin_amdgcn_global_load_async_to_lds_b128)
#define USE_ASYNC_LDS 1
#else
#define USE_ASYNC_LDS 0
#endif

#if USE_ASYNC_LDS
// Builtin signature (from the compiler diagnostic):
//   void (v4i addrspace(1)* src, v4i addrspace(3)* dst, imm offset, imm cpol)
typedef __attribute__((address_space(1))) v4i* gv4i_p;
typedef __attribute__((address_space(3))) v4i* lv4i_p;
__device__ __forceinline__ void wait_asynccnt0() {
#if __has_builtin(__builtin_amdgcn_s_wait_asynccnt)
  __builtin_amdgcn_s_wait_asynccnt(0);
#else
  asm volatile("s_wait_asynccnt 0x0" ::: "memory");
#endif
}
#endif

// ---------------------------------------------------------------------------
// Tiled transpose + f32 -> bf16 convert:  W (K x N, row-major f32)  ->
// Wt (N x K, row-major bf16).  Transposed storage makes the WMMA B-fragment
// (one column, 16 consecutive K values per lane) a contiguous 32-byte load.
// ---------------------------------------------------------------------------
__global__ __launch_bounds__(256) void transpose_convert(
    const float* __restrict__ W, bf16* __restrict__ Wt, int K, int N)
{
  __shared__ float tile[32][33];
  const int k0 = blockIdx.y * 32;
  const int n0 = blockIdx.x * 32;
  const int tx = threadIdx.x & 31;
  const int ty = threadIdx.x >> 5;          // 0..7
#pragma unroll
  for (int i = 0; i < 32; i += 8)
    tile[ty + i][tx] = W[(size_t)(k0 + ty + i) * N + n0 + tx];
  __syncthreads();
#pragma unroll
  for (int i = 0; i < 32; i += 8)
    Wt[(size_t)(n0 + ty + i) * K + k0 + tx] = (bf16)tile[tx][ty + i];
}

// ---------------------------------------------------------------------------
// Embedding gather: hbf[m, :] = bf16(emb[x[m], :]),  m = l*B + b.
// ---------------------------------------------------------------------------
__global__ __launch_bounds__(256) void gather_emb(
    const int* __restrict__ x, const float* __restrict__ emb,
    bf16* __restrict__ hbf)
{
  const int m = blockIdx.x;                 // 0..4095
  const int tok = x[m];
  const float* src = emb + (size_t)tok * N_E;
  bf16* dst = hbf + (size_t)m * N_E;
  for (int e = threadIdx.x; e < N_E; e += blockDim.x)
    dst[e] = (bf16)src[e];
}

// ---------------------------------------------------------------------------
// bf16 WMMA GEMM:  C[M,N] (f32) = A[M,K] (bf16, row-major)
//                               x Bt[N,K] (bf16, row-major = B transposed)
//                               (+ bias[N] if bias != nullptr)
// Double-buffered LDS; next tile is fetched (async-to-LDS on CDNA5) while the
// current tile feeds 16 WMMAs, so HBM/L2 latency hides behind matrix work.
// ---------------------------------------------------------------------------
__global__ __launch_bounds__(256) void gemm_bf16_wmma(
    const bf16*  __restrict__ A,
    const bf16*  __restrict__ Bt,
    const float* __restrict__ bias,
    float*       __restrict__ C,
    int M, int N, int K)
{
  __shared__ alignas(16) bf16 sA[2][BM * BK];
  __shared__ alignas(16) bf16 sB[2][BN * BK];

  const int tid  = threadIdx.x;
  const int lane = tid & 31;
  const int wave = tid >> 5;
  const int bm = blockIdx.y * BM;
  const int bn = blockIdx.x * BN;
  const int wm = (wave >> 1) * 32;          // wave row offset in block tile
  const int wn = (wave & 1) * 64;           // wave col offset in block tile

  const v8f vzero = {0.f, 0.f, 0.f, 0.f, 0.f, 0.f, 0.f, 0.f};
  v8f acc[2][4];
#pragma unroll
  for (int i = 0; i < 2; ++i)
#pragma unroll
    for (int j = 0; j < 4; ++j) acc[i][j] = vzero;

  // cooperative tile fill: 256 threads x 16B x 4 passes -> 128x64 bf16 tile
  const int ldrow = tid >> 3;               // 0..31
  const int ldcol = (tid & 7) * 8;          // 0,8,..,56 (bf16 elements)

  const int r  = lane & 15;                 // fragment row/col within 16
  const int hi = lane >> 4;                 // lane half select

#if USE_ASYNC_LDS
  auto issue_tile = [&](int buf, int kk) {
#pragma unroll
    for (int p = 0; p < 4; ++p) {
      const int row = p * 32 + ldrow;
      __builtin_amdgcn_global_load_async_to_lds_b128(
          (gv4i_p)&A[(size_t)(bm + row) * K + kk + ldcol],
          (lv4i_p)&sA[buf][row * BK + ldcol], 0, 0);
      __builtin_amdgcn_global_load_async_to_lds_b128(
          (gv4i_p)&Bt[(size_t)(bn + row) * K + kk + ldcol],
          (lv4i_p)&sB[buf][row * BK + ldcol], 0, 0);
    }
  };
  issue_tile(0, 0);
  wait_asynccnt0();
#else
  {
    v8bf ra[4], rb[4];
#pragma unroll
    for (int p = 0; p < 4; ++p) {
      const int row = p * 32 + ldrow;
      ra[p] = *reinterpret_cast<const v8bf*>(&A[(size_t)(bm + row) * K + ldcol]);
      rb[p] = *reinterpret_cast<const v8bf*>(&Bt[(size_t)(bn + row) * K + ldcol]);
    }
#pragma unroll
    for (int p = 0; p < 4; ++p) {
      const int row = p * 32 + ldrow;
      *reinterpret_cast<v8bf*>(&sA[0][row * BK + ldcol]) = ra[p];
      *reinterpret_cast<v8bf*>(&sB[0][row * BK + ldcol]) = rb[p];
    }
  }
#endif
  __syncthreads();

  for (int k0 = 0, buf = 0; k0 < K; k0 += BK, buf ^= 1) {
    const bool has_next = (k0 + BK) < K;

#if USE_ASYNC_LDS
    if (has_next) issue_tile(buf ^ 1, k0 + BK);
#else
    v8bf ra[4], rb[4];
    if (has_next) {
#pragma unroll
      for (int p = 0; p < 4; ++p) {
        const int row = p * 32 + ldrow;
        ra[p] = *reinterpret_cast<const v8bf*>(
            &A[(size_t)(bm + row) * K + k0 + BK + ldcol]);
        rb[p] = *reinterpret_cast<const v8bf*>(
            &Bt[(size_t)(bn + row) * K + k0 + BK + ldcol]);
      }
    }
#endif
    if (k0 + 2 * BK < K) {                  // warm L2/WGP$ two tiles ahead
      __builtin_prefetch(&A[(size_t)(bm + ldrow) * K + k0 + 2 * BK + ldcol]);
      __builtin_prefetch(&Bt[(size_t)(bn + ldrow) * K + k0 + 2 * BK + ldcol]);
    }

    const bf16* pAbase = &sA[buf][0];
    const bf16* pBbase = &sB[buf][0];
#pragma unroll
    for (int ks = 0; ks < 2; ++ks) {        // two K=32 WMMA steps per BK=64
      // A fragment (16x32 bf16): lanes 0-15 -> K 0..15 of row r,
      // lanes 16-31 -> K 16..31  (matches B fragment order)
      v16bf af[2];
#pragma unroll
      for (int i = 0; i < 2; ++i) {
        const bf16* pa = pAbase + (wm + i * 16 + r) * BK + ks * 32 + hi * 16;
        v8bf lo  = *reinterpret_cast<const v8bf*>(pa);
        v8bf hi8 = *reinterpret_cast<const v8bf*>(pa + 8);
        af[i] = __builtin_shufflevector(lo, hi8,
                 0,1,2,3,4,5,6,7,8,9,10,11,12,13,14,15);
      }
      // B fragment (32x16 bf16): lane half 0 -> K 0..15, half 1 -> K 16..31,
      // column = r.  Bt is N x K so this is one contiguous 32-byte load.
      v16bf bfr[4];
#pragma unroll
      for (int j = 0; j < 4; ++j) {
        const bf16* pb = pBbase + (wn + j * 16 + r) * BK + ks * 32 + hi * 16;
        bfr[j] = *reinterpret_cast<const v16bf*>(pb);
      }
#pragma unroll
      for (int i = 0; i < 2; ++i)
#pragma unroll
        for (int j = 0; j < 4; ++j)
          acc[i][j] = __builtin_amdgcn_wmma_f32_16x16x32_bf16(
              false, af[i], false, bfr[j], (short)0, acc[i][j], false, false);
    }

    if (has_next) {
#if USE_ASYNC_LDS
      wait_asynccnt0();                     // this wave's fills have landed
#else
#pragma unroll
      for (int p = 0; p < 4; ++p) {
        const int row = p * 32 + ldrow;
        *reinterpret_cast<v8bf*>(&sA[buf ^ 1][row * BK + ldcol]) = ra[p];
        *reinterpret_cast<v8bf*>(&sB[buf ^ 1][row * BK + ldcol]) = rb[p];
      }
#endif
      __syncthreads();                      // publish next buffer
    }
  }

  // Epilogue: C/D layout -> VGPR t holds M = t + 8*hi, N = r.
#pragma unroll
  for (int i = 0; i < 2; ++i) {
#pragma unroll
    for (int j = 0; j < 4; ++j) {
      const int cn  = bn + wn + j * 16 + r;
      const int cm0 = bm + wm + i * 16 + hi * 8;
      const float bv = bias ? bias[cn] : 0.0f;
#pragma unroll
      for (int t = 0; t < 8; ++t)
        C[(size_t)(cm0 + t) * N + cn] = acc[i][j][t] + bv;
    }
  }
}

// ---------------------------------------------------------------------------
// SRU recurrence.  f = sigmoid(u1 + vf*c + bias) depends on c, so the scan is
// inherently sequential in l.  One thread per (b, d); 8192 lanes total.
// The loads do NOT depend on c, so we software-pipeline in chunks of 4 steps:
// chunk i+1's 16 loads are issued before chunk i's ALU work, hiding latency.
// Writes h in f32 (next-layer residual) and bf16 (next GEMM A operand).
// ---------------------------------------------------------------------------
#define SCH 4
__global__ __launch_bounds__(256) void sru_scan(
    const float* __restrict__ U,       // (L, B, ncols, N_D) f32
    int ncols,                         // 4 (layer 0) or 3
    const float* __restrict__ residH,  // (L, B, N_D) f32, used when ncols==3
    const float* __restrict__ vcl,     // (2*N_D) f32: vf | vr
    const float* __restrict__ c0,      // (B, N_D) f32
    float* __restrict__ hOut,          // (L, B, N_D) f32
    bf16*  __restrict__ hBf,           // (L, B, N_D) bf16
    float* __restrict__ cOut)          // (B, N_D) f32
{
  const int idx = blockIdx.x * blockDim.x + threadIdx.x;
  if (idx >= BATCH * N_D) return;
  const int b = idx >> 10;
  const int d = idx & (N_D - 1);

  const float vf = vcl[d];
  const float vr = vcl[N_D + d];
  float c = c0[idx];

  float cu0[SCH], cu1[SCH], cu2[SCH], crt[SCH];

  auto load_chunk = [&](int l0, float* a0, float* a1, float* a2, float* art) {
#pragma unroll
    for (int t = 0; t < SCH; ++t) {
      const size_t row = (size_t)((l0 + t) * BATCH + b);
      const size_t ub  = row * ncols * N_D + d;
      a0[t] = U[ub];
      a1[t] = U[ub + N_D];
      a2[t] = U[ub + 2 * N_D];
      art[t] = (ncols == 4) ? U[ub + 3 * N_D] : residH[row * N_D + d];
    }
  };

  load_chunk(0, cu0, cu1, cu2, crt);

#pragma unroll 1
  for (int l0 = 0; l0 < SEQ_L; l0 += SCH) {
    float nu0[SCH], nu1[SCH], nu2[SCH], nrt[SCH];
    if (l0 + SCH < SEQ_L)
      load_chunk(l0 + SCH, nu0, nu1, nu2, nrt);   // in flight during ALU work

#pragma unroll
    for (int t = 0; t < SCH; ++t) {
      const float f = 1.0f / (1.0f + __expf(-(cu1[t] + vf * c + (-3.0f))));
      const float r = 1.0f / (1.0f + __expf(-(cu2[t] + vr * c)));
      c = f * c + (1.0f - f) * cu0[t];
      const float h = r * c + (1.0f - r) * crt[t];
      const size_t ho = (size_t)((l0 + t) * BATCH + b) * N_D + d;
      hOut[ho] = h;
      hBf[ho]  = (bf16)h;
    }
#pragma unroll
    for (int t = 0; t < SCH; ++t) {
      cu0[t] = nu0[t]; cu1[t] = nu1[t]; cu2[t] = nu2[t]; crt[t] = nrt[t];
    }
  }
  cOut[idx] = c;
}

// ---------------------------------------------------------------------------
// launcher
// ---------------------------------------------------------------------------
extern "C" void kernel_launch(void* const* d_in, const int* in_sizes, int n_in,
                              void* d_out, int out_size, void* d_ws, size_t ws_size,
                              hipStream_t stream)
{
  (void)in_sizes; (void)n_in; (void)out_size; (void)ws_size;

  const int*   x      = (const int*)  d_in[0];  // (L, B)
  const float* hidden = (const float*)d_in[1];  // (DEPTH, B, N_D)
  const float* emb    = (const float*)d_in[2];  // (VOCAB, N_E)
  const float* W0     = (const float*)d_in[3];  // (N_E, 4*N_D)
  const float* Wl     = (const float*)d_in[4];  // (DEPTH-1, N_D, 3*N_D)
  const float* vc     = (const float*)d_in[5];  // (DEPTH, 2*N_D)
  const float* outW   = (const float*)d_in[6];  // (N_D, VOCAB)
  const float* outb   = (const float*)d_in[7];  // (VOCAB,)

  float* out  = (float*)d_out;                          // (L*B, VOCAB)
  float* cOut = out + (size_t)M_TOT * VOCAB;            // (DEPTH, B, N_D)

  // workspace carve-up (~185 MB total)
  uintptr_t base = (uintptr_t)d_ws;
  size_t off = 0;
  auto take = [&](size_t bytes) -> uintptr_t {
    uintptr_t p = base + off;
    off = (off + bytes + 255) & ~(size_t)255;
    return p;
  };
  bf16*  Wt0 = (bf16*) take((size_t)(4 * N_D) * N_E * sizeof(bf16));
  bf16*  Wtl = (bf16*) take((size_t)(DEPTH - 1) * (3 * N_D) * N_D * sizeof(bf16));
  bf16*  Wto = (bf16*) take((size_t)VOCAB * N_D * sizeof(bf16));
  bf16*  hbf = (bf16*) take((size_t)M_TOT * N_E * sizeof(bf16));
  float* hf  = (float*)take((size_t)M_TOT * N_D * sizeof(float));
  float* U   = (float*)take((size_t)M_TOT * (4 * N_D) * sizeof(float));

  // 1) bf16 weight conversion, transposed to N x K for contiguous B-frags.
  //    All bf16 weights (~90 MB) fit in the 192 MB L2 -> GEMM rereads are
  //    L2-resident, not HBM traffic.
  transpose_convert<<<dim3((4 * N_D) / 32, N_E / 32), 256, 0, stream>>>(
      W0, Wt0, N_E, 4 * N_D);
  for (int l = 0; l < DEPTH - 1; ++l)
    transpose_convert<<<dim3((3 * N_D) / 32, N_D / 32), 256, 0, stream>>>(
        Wl + (size_t)l * N_D * (3 * N_D),
        Wtl + (size_t)l * (3 * N_D) * N_D, N_D, 3 * N_D);
  transpose_convert<<<dim3(VOCAB / 32, N_D / 32), 256, 0, stream>>>(
      outW, Wto, N_D, VOCAB);

  // 2) embedding gather -> bf16 activations
  gather_emb<<<M_TOT, 256, 0, stream>>>(x, emb, hbf);

  // 3) per-layer: GEMM (bf16 WMMA, f32 accum) then sequential SRU scan
  for (int layer = 0; layer < DEPTH; ++layer) {
    const int ncols = (layer == 0) ? 4 : 3;
    const int N = ncols * N_D;
    const bf16* Wt = (layer == 0) ? Wt0
                                  : Wtl + (size_t)(layer - 1) * (3 * N_D) * N_D;
    gemm_bf16_wmma<<<dim3(N / BN, M_TOT / BM), 256, 0, stream>>>(
        hbf, Wt, nullptr, U, M_TOT, N, N_E);
    sru_scan<<<(BATCH * N_D) / 256, 256, 0, stream>>>(
        U, ncols, hf, vc + (size_t)layer * 2 * N_D,
        hidden + (size_t)layer * BATCH * N_D,
        hf, hbf, cOut + (size_t)layer * BATCH * N_D);
  }

  // 4) output projection: 268 GFLOP, the dominant GEMM (N = 32000)
  gemm_bf16_wmma<<<dim3(VOCAB / BN, M_TOT / BM), 256, 0, stream>>>(
      hbf, Wto, outb, out, M_TOT, VOCAB, N_E);
}
